// WayfinderAttentionTorch_62440234549741
// MI455X (gfx1250) — matrix operand order repaired and checked by reading, not compiled, not run on hardware
//
#include <hip/hip_runtime.h>

typedef _Float16 v16h __attribute__((ext_vector_type(16)));
typedef _Float16 v8h  __attribute__((ext_vector_type(8)));
typedef _Float16 v4h  __attribute__((ext_vector_type(4)));
typedef float    v8f  __attribute__((ext_vector_type(8)));

#define TPB 32          // 1 wave per block, 1 token per wave
constexpr int Hc = 16, Tc = 2048, DHc = 64, Dc = 64;

__global__ __launch_bounds__(TPB) void wayfinder_attn(
    const float* __restrict__ q, const float* __restrict__ k,
    const float* __restrict__ v, const float* __restrict__ etb,
    const int* __restrict__ nidx, const int* __restrict__ etyp,
    float* __restrict__ out)
{
    __shared__ __align__(16) float    sKF[Dc * DHc];   // async-staged K rows, raw f32
    __shared__ __align__(32) _Float16 sK [Dc * DHc];   // K, row-major f16
    __shared__ __align__(32) _Float16 sVT[DHc * Dc];   // V, transposed f16
    __shared__ __align__(32) _Float16 sQ [DHc];        // q * 1/8, f16
    __shared__ __align__(32) float    sBM[Dc];         // bias or -1e30 (mask)
    __shared__ __align__(32) _Float16 sW [Dc];         // softmax weights f16
    __shared__ __align__(32) int      sIdx[Dc];        // clamped neighbor idx

    const int lane = threadIdx.x & 31;
    const int g    = blockIdx.x;               // flattened (h,t), B==1
    const int t    = g & (Tc - 1);
    const int bh   = g >> 11;                  // g / Tc
    const float* khead = k + (size_t)bh * Tc * DHc;
    const float* vhead = v + (size_t)bh * Tc * DHc;

    // ---------------- phase 0: idx / edge bias / mask / q ----------------
    const float4 tb = *reinterpret_cast<const float4*>(etb);  // bias table[1..4]
    #pragma unroll
    for (int hblk = 0; hblk < 2; ++hblk) {
        int d  = lane + 32 * hblk;
        int ii = nidx[(size_t)g * Dc + d];
        int ee = etyp[(size_t)g * Dc + d];
        bool valid = (ii >= 0) && (ii <= t);
        int  si = ii < 0 ? 0 : (ii > Tc - 1 ? Tc - 1 : ii);
        sIdx[d] = si;
        float bv = (ee == 1) ? tb.x : (ee == 2) ? tb.y :
                   (ee == 3) ? tb.z : (ee == 4) ? tb.w : 0.0f;
        sBM[d] = valid ? bv : -1.0e30f;
    }
    {
        const float2 qq =
            *reinterpret_cast<const float2*>(q + (size_t)g * DHc + 2 * lane);
        sQ[2 * lane + 0] = (_Float16)(qq.x * 0.125f);   // fold 1/sqrt(64)
        sQ[2 * lane + 1] = (_Float16)(qq.y * 0.125f);
    }
    __syncthreads();

    const int sub = lane >> 4;          // which of 2 rows per iteration
    const int eo  = (lane & 15) * 4;    // element offset within row

    // ---------- phase 1a: fire all K-row gathers via async-to-LDS --------
    // 32 x global_load_async_to_lds_b128 in flight (ASYNCcnt), no VGPR use.
    for (int it = 0; it < 32; ++it) {
        int row  = 2 * it + sub;
        int ridx = sIdx[row];
        unsigned lds_off = (unsigned)(uintptr_t)(&sKF[row * DHc + eo]);
        unsigned long long ga =
            (unsigned long long)(uintptr_t)(khead + (size_t)ridx * DHc + eo);
        asm volatile("global_load_async_to_lds_b128 %0, %1, off"
                     :: "v"(lds_off), "v"(ga) : "memory");
    }

    // ---------- phase 1b: V gather (register path, overlaps K asyncs) ----
    #pragma unroll 4
    for (int it = 0; it < 32; ++it) {
        int row  = 2 * it + sub;
        int ridx = sIdx[row];
        const float4 vv =
            *reinterpret_cast<const float4*>(vhead + (size_t)ridx * DHc + eo);
        sVT[(eo + 0) * Dc + row] = (_Float16)vv.x;
        sVT[(eo + 1) * Dc + row] = (_Float16)vv.y;
        sVT[(eo + 2) * Dc + row] = (_Float16)vv.z;
        sVT[(eo + 3) * Dc + row] = (_Float16)vv.w;
    }

    // ---------- phase 1c: drain asyncs, convert K f32 -> f16 -------------
    asm volatile("s_wait_asynccnt 0x0" ::: "memory");
    __syncthreads();
    #pragma unroll 4
    for (int it = 0; it < 32; ++it) {
        int row = 2 * it + sub;
        const float4 kv = *reinterpret_cast<const float4*>(&sKF[row * DHc + eo]);
        v4h kh;
        kh[0] = (_Float16)kv.x; kh[1] = (_Float16)kv.y;
        kh[2] = (_Float16)kv.z; kh[3] = (_Float16)kv.w;
        *reinterpret_cast<v4h*>(&sK[row * DHc + eo]) = kh;
    }
    __syncthreads();

    // ---------------- phase 2: scores = Kg * q (WMMA), C = bias+mask -----
    const int hi = sub;   // 0 for lanes 0..15, 1 for lanes 16..31
    v16h bq0 = *reinterpret_cast<const v16h*>(&sQ[ 0 + hi * 16]);
    v16h bq1 = *reinterpret_cast<const v16h*>(&sQ[32 + hi * 16]);

    v8f sc[4];
    #pragma unroll
    for (int mt = 0; mt < 4; ++mt) {
        int rowm = 16 * mt + (lane & 15);
        v8h p0 = *reinterpret_cast<const v8h*>(&sK[rowm * DHc +  0 + hi * 16]);
        v8h p1 = *reinterpret_cast<const v8h*>(&sK[rowm * DHc +  8 + hi * 16]);
        v8h p2 = *reinterpret_cast<const v8h*>(&sK[rowm * DHc + 32 + hi * 16]);
        v8h p3 = *reinterpret_cast<const v8h*>(&sK[rowm * DHc + 40 + hi * 16]);
        v16h a0 = __builtin_shufflevector(p0, p1, 0,1,2,3,4,5,6,7,8,9,10,11,12,13,14,15);
        v16h a1 = __builtin_shufflevector(p2, p3, 0,1,2,3,4,5,6,7,8,9,10,11,12,13,14,15);
        v8f c = *reinterpret_cast<const v8f*>(&sBM[16 * mt + 8 * hi]);
        c = __builtin_amdgcn_wmma_f32_16x16x32_f16(false, a0, false, bq0, (short)0, c, false, false);
        c = __builtin_amdgcn_wmma_f32_16x16x32_f16(false, a1, false, bq1, (short)0, c, false, false);
        sc[mt] = c;
    }

    // ---------------- softmax (lane-replicated halves + one swap) --------
    float mmax = -3.0e38f;
    #pragma unroll
    for (int mt = 0; mt < 4; ++mt)
        #pragma unroll
        for (int r = 0; r < 8; ++r) mmax = fmaxf(mmax, sc[mt][r]);
    mmax = fmaxf(mmax, __shfl_xor(mmax, 16, 32));

    float ssum = 0.0f;
    float ex[4][8];
    #pragma unroll
    for (int mt = 0; mt < 4; ++mt)
        #pragma unroll
        for (int r = 0; r < 8; ++r) {
            ex[mt][r] = __expf(sc[mt][r] - mmax);
            ssum += ex[mt][r];
        }
    ssum += __shfl_xor(ssum, 16, 32);
    const float inv = (mmax < -1.0e29f) ? 0.0f : 1.0f / fmaxf(ssum, 1e-20f);

    if ((lane & 15) == 0) {   // lanes 0 and 16 jointly cover all 64 d's
        #pragma unroll
        for (int mt = 0; mt < 4; ++mt)
            #pragma unroll
            for (int r = 0; r < 8; ++r)
                sW[16 * mt + 8 * hi + r] = (_Float16)(ex[mt][r] * inv);
    }
    __syncthreads();

    // ---------------- phase 3: out = V^T * w (WMMA) ----------------------
    v16h bw0 = *reinterpret_cast<const v16h*>(&sW[ 0 + hi * 16]);
    v16h bw1 = *reinterpret_cast<const v16h*>(&sW[32 + hi * 16]);
    float* orow = out + (size_t)g * DHc;
    #pragma unroll
    for (int mt = 0; mt < 4; ++mt) {
        int rowe = 16 * mt + (lane & 15);
        v8h p0 = *reinterpret_cast<const v8h*>(&sVT[rowe * Dc +  0 + hi * 16]);
        v8h p1 = *reinterpret_cast<const v8h*>(&sVT[rowe * Dc +  8 + hi * 16]);
        v8h p2 = *reinterpret_cast<const v8h*>(&sVT[rowe * Dc + 32 + hi * 16]);
        v8h p3 = *reinterpret_cast<const v8h*>(&sVT[rowe * Dc + 40 + hi * 16]);
        v16h a0 = __builtin_shufflevector(p0, p1, 0,1,2,3,4,5,6,7,8,9,10,11,12,13,14,15);
        v16h a1 = __builtin_shufflevector(p2, p3, 0,1,2,3,4,5,6,7,8,9,10,11,12,13,14,15);
        v8f o = {};
        o = __builtin_amdgcn_wmma_f32_16x16x32_f16(false, a0, false, bw0, (short)0, o, false, false);
        o = __builtin_amdgcn_wmma_f32_16x16x32_f16(false, a1, false, bw1, (short)0, o, false, false);
        if ((lane & 15) == 0) {
            #pragma unroll
            for (int r = 0; r < 8; ++r)
                orow[16 * mt + 8 * hi + r] = o[r];
        }
    }
}

extern "C" void kernel_launch(void* const* d_in, const int* in_sizes, int n_in,
                              void* d_out, int out_size, void* d_ws, size_t ws_size,
                              hipStream_t stream) {
    const float* q    = (const float*)d_in[0];
    const float* k    = (const float*)d_in[1];
    const float* v    = (const float*)d_in[2];
    const float* etb  = (const float*)d_in[3];
    const int*   nidx = (const int*)d_in[4];
    const int*   etyp = (const int*)d_in[5];
    float* out = (float*)d_out;

    const int tokens = Hc * Tc;          // B==1, one token per block (one wave)
    dim3 grid(tokens), block(TPB);
    wayfinder_attn<<<grid, block, 0, stream>>>(q, k, v, etb, nidx, etyp, out);
}
